// GptOssExpertsAsLinear_57114475102290
// MI455X (gfx1250) — matrix operand order repaired and checked by reading, not compiled, not run on hardware
//
#include <hip/hip_runtime.h>
#include <hip/hip_bf16.h>
#include <math.h>

// ---------------- problem constants ----------------
#define TOK   4096
#define HDIM  1024
#define DDIM  1024
#define TWO_D 2048
#define NEXP  8
#define TOPK  2
#define ALPHA 1.702f
#define LIMIT 7.0f

// ---------------- tiling ----------------
#define MT    64            // token tile per block
#define KBLK  128           // K elements staged per stage
#define NKB   (HDIM / KBLK) // 8 k-blocks
#define LSTR  136           // padded LDS row stride (elements, 272B = 17x16B)
#define NTH   512           // 16 waves: 4(M) x 4(N)

// ---------------- TDM availability ----------------
#if defined(__has_builtin)
#  if __has_builtin(__builtin_amdgcn_tensor_load_to_lds)
#    define HAS_TDM 1
#  endif
#endif
#ifndef HAS_TDM
#  define HAS_TDM 0
#endif

typedef __attribute__((ext_vector_type(16))) __bf16        bf16x16;
typedef __attribute__((ext_vector_type(8)))  float         f32x8;
typedef __attribute__((ext_vector_type(4)))  float         f32x4;
typedef __attribute__((ext_vector_type(4)))  unsigned int  u32x4;
typedef __attribute__((ext_vector_type(2)))  unsigned int  u32x2;
typedef __attribute__((ext_vector_type(4)))  unsigned int  v4u;
typedef __attribute__((ext_vector_type(8)))  int           v8i;
typedef __attribute__((ext_vector_type(4)))  int           v4i;

union FragU { bf16x16 v; u32x4 q[2]; };
union Pk4   { __bf16 h[4]; u32x2 v; };

// A-operand (16x32 bf16): lane m = l&15; halves at kb=(l>>4)*8 and kb+16
__device__ __forceinline__ bf16x16 load_fragA(const __bf16* base, int stride, int lane) {
  const __bf16* p = base + (lane & 15) * stride + ((lane >> 4) << 3);
  FragU u;
  u.q[0] = *(const u32x4*)(p);
  u.q[1] = *(const u32x4*)(p + 16);
  return u.v;
}
// B-operand (32x16 bf16): lane n = l&15; halves at kb=(l>>4)*8 and kb+16
__device__ __forceinline__ bf16x16 load_fragB(const __bf16* base, int stride, int lane) {
  const __bf16* p = base + (lane & 15) * stride + ((lane >> 4) << 3);
  FragU u;
  u.q[0] = *(const u32x4*)(p);
  u.q[1] = *(const u32x4*)(p + 16);
  return u.v;
}

// ---- TDM: copy 64 rows x 128 bf16 (row stride 1024 elem) -> LDS rows of LSTR ----
// pad_enable, interval=64 dwords (code 5), amount=4 dwords (code 3) => LSTR=136
__device__ __forceinline__ void tdm_stage_B(const __bf16* gsrc, __bf16* ldsDst, int tid) {
#if HAS_TDM
  if ((tid >> 5) == 0) {
    unsigned long long ga = (unsigned long long)(uintptr_t)gsrc;
    v4u g0;
    g0[0] = 1u;                                            // count=1, user mode
    g0[1] = (unsigned)(uintptr_t)ldsDst;                   // lds byte offset
    g0[2] = (unsigned)(ga & 0xFFFFFFFFu);                  // global_addr lo
    g0[3] = (unsigned)((ga >> 32) & 0x01FFFFFFu) | (2u << 30); // addr hi + type=2
    v8i g1;
    g1[0] = (1 << 16)            // data_size = 2 bytes
          | (1 << 20)            // pad_enable
          | (5 << 22)            // pad_interval: 64 dwords
          | (3 << 25);           // pad_amount: 4 dwords (16B)
    g1[1] = (int)((1024u & 0xFFFFu) << 16);                // tensor_dim0[15:0] @ [63:48]
    g1[2] = (int)(((1024u >> 16) & 0xFFFFu) | (64u << 16));// dim0 hi | tensor_dim1 lo
    g1[3] = (int)(0u | (128u << 16));                      // dim1 hi | tile_dim0=128
    g1[4] = (int)(64u);                                    // tile_dim1=64, tile_dim2=0
    g1[5] = (int)(1024u);                                  // tensor_dim0_stride lo32
    g1[6] = 0;                                             // stride hi | dim1_stride lo
    g1[7] = 0;
    v4i g2 = (v4i)0, g3 = (v4i)0;
#  if defined(__clang_major__) && (__clang_major__ >= 23)
    v8i g4 = (v8i)0;
    __builtin_amdgcn_tensor_load_to_lds(g0, g1, g2, g3, g4, 0);
#  else
    __builtin_amdgcn_tensor_load_to_lds(g0, g1, g2, g3, 0);
#  endif
  }
#else
  // fallback: all 512 threads, vector copy (16 bf16 each)
  const int n  = tid >> 3;
  const int ks = (tid & 7) << 4;
  const u32x4* s = (const u32x4*)(gsrc + (size_t)n * 1024 + ks);
  u32x4* d = (u32x4*)(ldsDst + n * LSTR + ks);
  d[0] = s[0]; d[1] = s[1];
#endif
}

__device__ __forceinline__ void tdm_wait(int tid) {
#if HAS_TDM
#  if __has_builtin(__builtin_amdgcn_s_wait_tensorcnt)
  if ((tid >> 5) == 0) __builtin_amdgcn_s_wait_tensorcnt(0);
#  endif
#endif
}

// ---------------- kernel: zero output ----------------
__global__ void moe_zero_kernel(float* __restrict__ out, int n) {
  int i = blockIdx.x * blockDim.x + threadIdx.x;
  int stride = gridDim.x * blockDim.x;
  for (; i < n; i += stride) out[i] = 0.0f;
}

// ---------------- kernel: hidden f32 -> bf16 ----------------
__global__ void moe_hsconv_kernel(const float* __restrict__ src, __bf16* __restrict__ dst, int n4) {
  int i = blockIdx.x * blockDim.x + threadIdx.x;
  int stride = gridDim.x * blockDim.x;
  for (; i < n4; i += stride) {
    f32x4 v = *(const f32x4*)(src + (size_t)i * 4);
    Pk4 o;
    o.h[0] = (__bf16)v.x; o.h[1] = (__bf16)v.y; o.h[2] = (__bf16)v.z; o.h[3] = (__bf16)v.w;
    *(u32x2*)(dst + (size_t)i * 4) = o.v;
  }
}

// ---------------- kernel: weight f32 [K][N] -> bf16 [N][K] (per expert) ----------------
__global__ void moe_wconv_kernel(const float* __restrict__ src, __bf16* __restrict__ dst,
                                 int K, int N) {
  __shared__ float tile[32][33];
  const int e  = blockIdx.z;
  const float* s = src + (size_t)e * K * N;
  __bf16*      d = dst + (size_t)e * N * K;
  const int k0 = blockIdx.y * 32;
  const int n0 = blockIdx.x * 32;
  const int r  = threadIdx.x >> 3;   // 0..31
  const int c  = threadIdx.x & 7;    // 0..7 -> 4 cols each
  f32x4 v = *(const f32x4*)(s + (size_t)(k0 + r) * N + n0 + c * 4);
  tile[r][c * 4 + 0] = v.x; tile[r][c * 4 + 1] = v.y;
  tile[r][c * 4 + 2] = v.z; tile[r][c * 4 + 3] = v.w;
  __syncthreads();
  __bf16* drow = d + (size_t)(n0 + r) * K + k0;
  Pk4 o;
  o.h[0] = (__bf16)tile[c * 4 + 0][r]; o.h[1] = (__bf16)tile[c * 4 + 1][r];
  o.h[2] = (__bf16)tile[c * 4 + 2][r]; o.h[3] = (__bf16)tile[c * 4 + 3][r];
  *(u32x2*)(drow + c * 4) = o.v;
}

// ---------------- kernel: deterministic router compaction ----------------
__global__ void moe_router_kernel(const int* __restrict__ ridx,
                                  const float* __restrict__ rw,
                                  int* __restrict__ counts,
                                  int* __restrict__ list,
                                  float* __restrict__ wlist) {
  const int lane = threadIdx.x & 31;
  const int e    = threadIdx.x >> 5;
  if (e >= NEXP) return;
  int base = 0;
  for (int t0 = 0; t0 < TOK; t0 += 32) {
    const int t  = t0 + lane;
    const int i0 = ridx[t * TOPK + 0];
    const int i1 = ridx[t * TOPK + 1];
    const int cnt = (i0 == e) + (i1 == e);
    const bool act = cnt > 0;
    const unsigned mask = (unsigned)__ballot(act);
    const int pre = __popc(mask & ((1u << lane) - 1u));
    if (act) {
      const int pos = base + pre;
      list [e * TOK + pos] = t;
      wlist[e * TOK + pos] = rw[t * NEXP + e] * (float)cnt;
    }
    base += __popc(mask);
  }
  if (lane == 0) counts[e] = base;
}

// ---------------- kernel: fused expert (gate_up -> GLU -> down) ----------------
__global__ __launch_bounds__(NTH, 1)
void moe_expert_kernel(const __bf16* __restrict__ hsbf,   // [TOK][H] bf16
                       const __bf16* __restrict__ WguT,   // [E][2D][H] bf16 (transposed)
                       const float*  __restrict__ bgu,    // [E][2D]
                       const __bf16* __restrict__ WdT,    // [E][H][D] bf16 (transposed)
                       const float*  __restrict__ bd,     // [E][H]
                       const int*    __restrict__ counts,
                       const int*    __restrict__ list,
                       const float*  __restrict__ wlist,
                       float* __restrict__ out) {         // [TOK][H]
  const int e    = blockIdx.y;
  const int ne   = counts[e];
  const int tile = blockIdx.x;
  if (tile * MT >= ne) return;                 // block-uniform: EXEC stays full

  __shared__ __align__(16) __bf16 As [NKB * MT * LSTR];   // 136 KB, blocked [kb][m][136]
  __shared__ __align__(16) __bf16 Mid[NKB * MT * LSTR];   // 136 KB
  __shared__ __align__(16) __bf16 Bs [2][64 * LSTR];      // 34 KB double buffer
  __shared__ int   toks[MT];
  __shared__ float wv  [MT];

  const int tid   = threadIdx.x;
  const int lane  = tid & 31;
  const int wave  = tid >> 5;
  const int waveM = wave >> 2;                 // 0..3
  const int waveN = wave & 3;                  // 0..3

  if (tid < MT) {
    const int g = tile * MT + tid;
    if (g < ne) { toks[tid] = list[e * TOK + g]; wv[tid] = wlist[e * TOK + g]; }
    else        { toks[tid] = 0;                wv[tid] = 0.0f; }
  }
  __syncthreads();

  // ---- stage A (bf16, blocked layout): row = tid>>3, k-block = tid&7 ----
  {
    const int row = tid >> 3;
    const int kb  = tid & 7;
    const u32x4* src = (const u32x4*)(hsbf + (size_t)toks[row] * HDIM + kb * KBLK);
    u32x4* dst = (u32x4*)&As[(kb * MT + row) * LSTR];
    #pragma unroll
    for (int i = 0; i < 16; ++i) dst[i] = src[i];
  }

  const __bf16* WguTe = WguT + (size_t)e * TWO_D * HDIM;
  const __bf16* WdTe  = WdT  + (size_t)e * HDIM * DDIM;

  // ================= phase 1: gate_up + GLU =================
  {
    const int nStages = (TWO_D / 64) * NKB;    // 256
    tdm_stage_B(WguTe, &Bs[0][0], tid);        // prologue: stage 0 -> buf 0
    tdm_wait(tid);
    __syncthreads();
    int stage = 0;
    for (int nc = 0; nc < TWO_D; nc += 64) {
      f32x8 acc = {};
      for (int kb = 0; kb < NKB; ++kb) {
        const int buf = stage & 1;
        const int ns  = stage + 1;
        if (ns < nStages) {                    // prefetch next stage into other buffer
          const int nnc = (ns >> 3) * 64;
          const int nkk = (ns & 7) * KBLK;
          tdm_stage_B(WguTe + (size_t)nnc * HDIM + nkk, &Bs[buf ^ 1][0], tid);
        }
        const __bf16* abase = &As[(kb * MT + waveM * 16) * LSTR];
        const __bf16* bbase = &Bs[buf][(waveN * 16) * LSTR];
        #pragma unroll
        for (int ks = 0; ks < KBLK; ks += 32) {
          bf16x16 a = load_fragA(abase + ks, LSTR, lane);
          bf16x16 b = load_fragB(bbase + ks, LSTR, lane);
          acc = __builtin_amdgcn_wmma_f32_16x16x32_bf16(false, a, false, b,
                                                        (short)0, acc, false, false);
        }
        tdm_wait(tid);
        __syncthreads();
        ++stage;
      }
      // ---- bias + clamped GLU; interleaved gate/up live on adjacent lanes ----
      const int gcol = nc + waveN * 16 + (lane & 15);
      const float bias = bgu[(size_t)e * TWO_D + gcol];
      const int rowBase = waveM * 16 + ((lane >> 4) << 3);
      #pragma unroll
      for (int r = 0; r < 8; ++r) {
        float v = acc[r] + bias;
        float o = __shfl_xor(v, 1);
        float gate = (lane & 1) ? o : v;
        float up   = (lane & 1) ? v : o;
        gate = fminf(gate, LIMIT);
        up   = fminf(fmaxf(up, -LIMIT), LIMIT);
        const float sig  = 1.0f / (1.0f + __expf(-ALPHA * gate));
        const float midv = (up + 1.0f) * gate * sig;
        if ((lane & 1) == 0) {
          const int cm = gcol >> 1;
          Mid[((cm >> 7) * MT + rowBase + r) * LSTR + (cm & 127)] = (__bf16)midv;
        }
      }
    }
  }

  // ================= phase 2: down projection =================
  {
    const int nStages = (HDIM / 64) * NKB;     // 128
    tdm_stage_B(WdTe, &Bs[0][0], tid);
    tdm_wait(tid);
    __syncthreads();                            // also orders Mid writes before reads
    int stage = 0;
    for (int nc = 0; nc < HDIM; nc += 64) {
      f32x8 acc = {};
      for (int kb = 0; kb < NKB; ++kb) {
        const int buf = stage & 1;
        const int ns  = stage + 1;
        if (ns < nStages) {
          const int nnc = (ns >> 3) * 64;
          const int nkk = (ns & 7) * KBLK;
          tdm_stage_B(WdTe + (size_t)nnc * DDIM + nkk, &Bs[buf ^ 1][0], tid);
        }
        const __bf16* abase = &Mid[(kb * MT + waveM * 16) * LSTR];
        const __bf16* bbase = &Bs[buf][(waveN * 16) * LSTR];
        #pragma unroll
        for (int ks = 0; ks < KBLK; ks += 32) {
          bf16x16 a = load_fragA(abase + ks, LSTR, lane);
          bf16x16 b = load_fragB(bbase + ks, LSTR, lane);
          acc = __builtin_amdgcn_wmma_f32_16x16x32_bf16(false, a, false, b,
                                                        (short)0, acc, false, false);
        }
        tdm_wait(tid);
        __syncthreads();
        ++stage;
      }
      // ---- weighted scatter-accumulate: out[tok] += (y + bd) * w ----
      const int col = nc + waveN * 16 + (lane & 15);
      const float bias = bd[(size_t)e * HDIM + col];
      const int rowBase = waveM * 16 + ((lane >> 4) << 3);
      #pragma unroll
      for (int r = 0; r < 8; ++r) {
        const int row = rowBase + r;
        const float val = (acc[r] + bias) * wv[row];
        atomicAdd(&out[(size_t)toks[row] * HDIM + col], val);
      }
    }
  }
}

// ---------------- workspace layout (bytes) ----------------
#define WS_COUNTS 0
#define WS_LIST   256
#define WS_WLIST  (WS_LIST + (size_t)NEXP * TOK * 4)
#define WS_HSBF   (WS_WLIST + (size_t)NEXP * TOK * 4)
#define WS_WGUT   (WS_HSBF + (size_t)TOK * HDIM * 2)
#define WS_WDT    (WS_WGUT + (size_t)NEXP * TWO_D * HDIM * 2)
#define WS_END    (WS_WDT + (size_t)NEXP * HDIM * DDIM * 2)

// ---------------- host launcher ----------------
extern "C" void kernel_launch(void* const* d_in, const int* in_sizes, int n_in,
                              void* d_out, int out_size, void* d_ws, size_t ws_size,
                              hipStream_t stream) {
  const float* hs   = (const float*)d_in[0];
  const int*   ridx = (const int*)  d_in[1];
  const float* rw   = (const float*)d_in[2];
  const float* wgu  = (const float*)d_in[3];
  const float* bgu  = (const float*)d_in[4];
  const float* wd   = (const float*)d_in[5];
  const float* bd   = (const float*)d_in[6];
  float* out = (float*)d_out;

  if (ws_size < WS_END) return;                // need ~57 MB of scratch

  char* ws = (char*)d_ws;
  int*    counts = (int*)   (ws + WS_COUNTS);
  int*    list   = (int*)   (ws + WS_LIST);
  float*  wlist  = (float*) (ws + WS_WLIST);
  __bf16* hsbf   = (__bf16*)(ws + WS_HSBF);
  __bf16* WguT   = (__bf16*)(ws + WS_WGUT);
  __bf16* WdT    = (__bf16*)(ws + WS_WDT);

  moe_zero_kernel<<<512, 256, 0, stream>>>(out, out_size);
  moe_router_kernel<<<1, 256, 0, stream>>>(ridx, rw, counts, list, wlist);
  moe_hsconv_kernel<<<2048, 256, 0, stream>>>(hs, hsbf, TOK * HDIM / 4);
  {
    dim3 g1(TWO_D / 32, HDIM / 32, NEXP);      // gate_up: [K=H][N=2D] -> [N][K]
    moe_wconv_kernel<<<g1, 256, 0, stream>>>(wgu, WguT, HDIM, TWO_D);
    dim3 g2(HDIM / 32, DDIM / 32, NEXP);       // down: [K=D][N=H] -> [N][K]
    moe_wconv_kernel<<<g2, 256, 0, stream>>>(wd, WdT, DDIM, HDIM);
  }
  dim3 grid((TOK + MT - 1) / MT, NEXP);
  moe_expert_kernel<<<grid, NTH, 0, stream>>>(hsbf, WguT, bgu, WdT, bd,
                                              counts, list, wlist, out);
}
